// RNN_Stack_23398981828886
// MI455X (gfx1250) — matrix operand order repaired and checked by reading, not compiled, not run on hardware
//
#include <hip/hip_runtime.h>
#include <hip/hip_bf16.h>

// ---------------------------------------------------------------------------
// Problem constants (match the JAX reference)
// ---------------------------------------------------------------------------
#define T_STEPS 512
#define B_SZ    256
#define I_SZ    64
#define N_SZ    512
#define C_SZ    10
#define L_SZ    3

#define KB_N    (N_SZ / 32)   // 16 k-blocks of 32 for N-K GEMMs
#define KB_I    (I_SZ / 32)   // 2  k-blocks for the input projection
#define NB_N    (N_SZ / 16)   // 32 column tiles of 16
#define NSPLIT  4             // N-chunks (128 cols each) -> gridDim.x
#define MTILES  16            // batch tiles of 16 rows  -> gridDim.y
#define WAVES   8             // one 16-col tile per wave
#define THREADS 256           // 8 waves (wave32)
#define MEMBERS (L_SZ * NSPLIT)          // barrier group per batch tile
#define ROUNDS  (T_STEPS + L_SZ - 1)     // pipelined rounds

// LDS carve (dynamic shared): weights resident for the whole scan
#define LDS_WREC   0                       // 128 KB  Whh[l] chunk
#define LDS_WFEED  (128 * 1024)            // 128 KB  Whi[l-1] chunk
#define LDS_OLD    (256 * 1024)            // 16 KB   h_l(t-1) rows
#define LDS_NEW    (272 * 1024)            // 16 KB   h_{l-1}(t) rows
#define LDS_X      (288 * 1024)            // 2 KB    data[t] tile
#define LDS_BYTES  (290 * 1024)

typedef __attribute__((ext_vector_type(16))) __bf16 v16bf;
typedef __attribute__((ext_vector_type(8)))  float  v8f;

// ---------------------------------------------------------------------------
// Pack a row-major f32 weight W[K][N_SZ] into a bf16 B-operand layout whose k order matches this file's A loads.
// Per 32(K)x16(N) tile: lane = (n%16) + 16*k_half, 16 contiguous bf16 per
// lane -> consumers do 2x b128 loads.  Tile order (nb*KB + kb) streams the
// k-loop contiguously and makes each 8-tile column chunk a flat 128 KB blob.
// ---------------------------------------------------------------------------
__global__ void pack_b_kernel(const float* __restrict__ W, __bf16* __restrict__ dst,
                              int K, int zero_diag) {
    const int KBv = K / 32;
    const int tid = blockIdx.x * blockDim.x + threadIdx.x;
    const int total = KBv * NB_N * 32;
    if (tid >= total) return;
    const int lane = tid & 31;
    const int tile = tid >> 5;
    const int kb = tile % KBv;
    const int nb = tile / KBv;
    const int n = nb * 16 + (lane & 15);
    const int kbase = kb * 32 + (lane >> 4) * 16;
    __bf16* out = dst + ((size_t)(nb * KBv + kb) * 32 + lane) * 16;
#pragma unroll
    for (int j = 0; j < 16; ++j) {
        const int k = kbase + j;
        float v = W[(size_t)k * N_SZ + n];
        if (zero_diag && k == n) v = 0.0f;     // forward() zeroes recurrent diag
        out[j] = (__bf16)v;
    }
}

__global__ void f32_to_bf16_kernel(const float* __restrict__ src,
                                   __bf16* __restrict__ dst, int n) {
    const int i = blockIdx.x * blockDim.x + threadIdx.x;
    if (i < n) dst[i] = (__bf16)src[i];
}

// h_l(-1) is read from parity buffer 1 at t=0; also zero the barrier words.
__global__ void init_state_kernel(const float* __restrict__ h0,
                                  __bf16* __restrict__ hbf,
                                  unsigned* __restrict__ bar, int n) {
    const int i = blockIdx.x * blockDim.x + threadIdx.x;
    if (i < 64) bar[i] = 0u;
    if (i < n) hbf[(size_t)n + i] = (__bf16)h0[i];   // buffer parity 1
}

// ---------------------------------------------------------------------------
// Async global->LDS staging (ASYNCcnt path, no VGPR round trip).
// ---------------------------------------------------------------------------
__device__ __forceinline__ void stage_async(const void* gsrc, void* lds,
                                            int bytes, int tid) {
    unsigned ldsAddr = (unsigned)(uintptr_t)lds + (unsigned)tid * 16u;
    const char* g = (const char*)gsrc + (size_t)tid * 16u;
    const int iters = bytes / (16 * THREADS);
    for (int j = 0; j < iters; ++j) {
        asm volatile("global_load_async_to_lds_b128 %0, %1, off"
                     :: "v"(ldsAddr), "v"(g) : "memory");
        ldsAddr += 16u * THREADS;
        g += (size_t)16 * THREADS;
    }
}

__device__ __forceinline__ void stage2k(const void* gsrc, void* lds, int tid) {
    if (tid < 128) {
        unsigned ldsAddr = (unsigned)(uintptr_t)lds + (unsigned)tid * 16u;
        const char* g = (const char*)gsrc + (size_t)tid * 16u;
        asm volatile("global_load_async_to_lds_b128 %0, %1, off"
                     :: "v"(ldsAddr), "v"(g) : "memory");
    }
}

__device__ __forceinline__ void stage_wait() {
    asm volatile("s_wait_asynccnt 0x0" ::: "memory");
}

// ---------------------------------------------------------------------------
// 16x16x512 WMMA tile with two accumulator chains; A and B both from LDS.
// ---------------------------------------------------------------------------
__device__ __forceinline__ void wmma_lds2(const __bf16* lds_state,
                                          const __bf16* lds_w,
                                          int wv, int lane, v8f& c0, v8f& c1) {
    const int half = lane >> 4;
    const int mrow = lane & 15;
#pragma unroll
    for (int kb = 0; kb < KB_N; kb += 2) {
        v16bf a0, b0, a1, b1;
        const __bf16* ar = lds_state + mrow * N_SZ + kb * 32 + half * 16;
        ((uint4*)&a0)[0] = *(const uint4*)(ar);
        ((uint4*)&a0)[1] = *(const uint4*)(ar + 8);
        ((uint4*)&a1)[0] = *(const uint4*)(ar + 32);
        ((uint4*)&a1)[1] = *(const uint4*)(ar + 40);
        const uint4* bp = (const uint4*)(lds_w + ((size_t)(wv * KB_N + kb) * 32 + lane) * 16);
        ((uint4*)&b0)[0] = bp[0];
        ((uint4*)&b0)[1] = bp[1];
        ((uint4*)&b1)[0] = bp[64];
        ((uint4*)&b1)[1] = bp[65];
        c0 = __builtin_amdgcn_wmma_f32_16x16x32_bf16(false, a0, false, b0,
                                                     (short)0, c0, false, false);
        c1 = __builtin_amdgcn_wmma_f32_16x16x32_bf16(false, a1, false, b1,
                                                     (short)0, c1, false, false);
    }
}

// ---------------------------------------------------------------------------
// Layer-pipelined persistent kernel. grid = (NSPLIT, MTILES, L_SZ).
// Round R: layer l computes timestep t = R - l.  One group barrier per round
// (members = 3*NSPLIT WGs sharing a batch tile).  Weights live in LDS for the
// whole scan; the wave's own f32 recurrent tile lives in registers.
// ---------------------------------------------------------------------------
__global__ __launch_bounds__(THREADS, 1)
void rnn_stack_kernel(const __bf16* __restrict__ data_bf,   // [T][B][I]
                      const __bf16* __restrict__ pkWin,     // packed, K=64
                      const __bf16* __restrict__ pkWhh,     // [3] packed, K=512
                      const __bf16* __restrict__ pkWhi,     // [2] packed, K=512
                      const float*  __restrict__ h0,        // [3][256][512]
                      const float*  __restrict__ b_in,      // [512]
                      const float*  __restrict__ b_hh,      // [3][512]
                      const float*  __restrict__ b_hi,      // [2][512]
                      __bf16*       __restrict__ hbf,       // [2][3][256][512]
                      float*        __restrict__ h2fin,     // [256][512]
                      unsigned*     __restrict__ bar) {
    extern __shared__ __align__(16) char smem[];
    __bf16* ldsWrec  = (__bf16*)(smem + LDS_WREC);
    __bf16* ldsWfeed = (__bf16*)(smem + LDS_WFEED);
    __bf16* ldsOld   = (__bf16*)(smem + LDS_OLD);
    __bf16* ldsNew   = (__bf16*)(smem + LDS_NEW);
    __bf16* ldsX     = (__bf16*)(smem + LDS_X);

    const int tid   = threadIdx.x;
    const int lane  = tid & 31;
    const int wave  = tid >> 5;
    const int chunk = blockIdx.x;                 // 128-col chunk 0..3
    const int mtile = blockIdx.y;
    const int l     = blockIdx.z;                 // layer of this WG
    const int nb    = chunk * WAVES + wave;       // 16-col tile 0..31
    const int ncol  = nb * 16 + (lane & 15);
    const int half  = lane >> 4;
    const int mrow  = lane & 15;

    unsigned* bcnt = bar + mtile;
    unsigned* bgen = bar + 16 + mtile;

    const size_t layerStride = (size_t)B_SZ * N_SZ;
    const size_t copyStride  = (size_t)L_SZ * layerStride;
    const size_t chunkElems  = (size_t)WAVES * KB_N * 32 * 16;   // 65536 (128KB)
    const size_t mrowOff     = (size_t)mtile * 16 * N_SZ;

    // ---- preload this WG's weight chunks into LDS (once) ----
    stage_async(pkWhh + (size_t)l * N_SZ * N_SZ + (size_t)chunk * chunkElems,
                ldsWrec, 128 * 1024, tid);
    if (l > 0)
        stage_async(pkWhi + (size_t)(l - 1) * N_SZ * N_SZ + (size_t)chunk * chunkElems,
                    ldsWfeed, 128 * 1024, tid);
    stage_wait();
    __syncthreads();

    // ---- own recurrent f32 tile in registers ----
    v8f hprev;
    {
        const float* h0l = h0 + (size_t)l * layerStride;
#pragma unroll
        for (int r = 0; r < 8; ++r)
            hprev[r] = h0l[(size_t)(mtile * 16 + half * 8 + r) * N_SZ + ncol];
    }

    const float bias = b_in[ncol] + b_hh[(size_t)l * N_SZ + ncol] +
                       (l > 0 ? b_hi[(size_t)(l - 1) * N_SZ + ncol] : 0.0f);
    const float inv_tau = 0.5f;                 // 1/TAU, TAU = 2
    const float alpha   = 1.0f - inv_tau;

    for (int R = 0; R < ROUNDS; ++R) {
        const int t = R - l;
        if (t >= 0 && t < T_STEPS) {
            const int rb = t & 1;               // parity buffer written at t
            const int pb = rb ^ 1;              // parity buffer of t-1
            // stage inputs (async -> LDS)
            stage_async(hbf + (size_t)pb * copyStride + (size_t)l * layerStride + mrowOff,
                        ldsOld, 16 * 1024, tid);
            if (l > 0)
                stage_async(hbf + (size_t)rb * copyStride + (size_t)(l - 1) * layerStride + mrowOff,
                            ldsNew, 16 * 1024, tid);
            stage2k(data_bf + ((size_t)t * B_SZ + (size_t)mtile * 16) * I_SZ, ldsX, tid);
            if (t + 1 < T_STEPS)                 // warm next data tile
                __builtin_prefetch(data_bf + ((size_t)(t + 1) * B_SZ + (size_t)mtile * 16) * I_SZ
                                       + (size_t)tid * 8, 0, 3);
            stage_wait();
            __syncthreads();

            // xin = (data[t] @ Win) tile; Win B-tiles from global (tiny)
            v8f c0 = {}, c1 = {}, c2 = {}, c3 = {};
            {
                v16bf a0, b0, a1, b1;
                const __bf16* ar = ldsX + mrow * I_SZ + half * 16;
                ((uint4*)&a0)[0] = *(const uint4*)(ar);
                ((uint4*)&a0)[1] = *(const uint4*)(ar + 8);
                ((uint4*)&a1)[0] = *(const uint4*)(ar + 32);
                ((uint4*)&a1)[1] = *(const uint4*)(ar + 40);
                const uint4* bp = (const uint4*)(pkWin + ((size_t)(nb * KB_I) * 32 + lane) * 16);
                ((uint4*)&b0)[0] = bp[0];
                ((uint4*)&b0)[1] = bp[1];
                ((uint4*)&b1)[0] = bp[64];
                ((uint4*)&b1)[1] = bp[65];
                c0 = __builtin_amdgcn_wmma_f32_16x16x32_bf16(false, a0, false, b0,
                                                             (short)0, c0, false, false);
                c1 = __builtin_amdgcn_wmma_f32_16x16x32_bf16(false, a1, false, b1,
                                                             (short)0, c1, false, false);
            }
            // recurrent + feed-in GEMMs, weights from LDS
            wmma_lds2(ldsOld, ldsWrec, wave, lane, c0, c1);
            if (l > 0)
                wmma_lds2(ldsNew, ldsWfeed, wave, lane, c2, c3);
            v8f acc = (c0 + c1) + (c2 + c3);

            // epilogue: leaky integration + LeakyReLU(0.01)
            __bf16* hbNew = hbf + (size_t)rb * copyStride + (size_t)l * layerStride;
#pragma unroll
            for (int r = 0; r < 8; ++r) {
                const size_t idx = (size_t)(mtile * 16 + half * 8 + r) * N_SZ + ncol;
                float v = alpha * hprev[r] + (acc[r] + bias) * inv_tau;
                v = v > 0.0f ? v : 0.01f * v;
                hprev[r] = v;
                hbNew[idx] = (__bf16)v;
                if (l == 2 && t == T_STEPS - 1) h2fin[idx] = v;
            }
        }

        // ---- one barrier per round across the 12 WGs of this batch tile ----
        __threadfence();
        __syncthreads();
        if (tid == 0) {
            unsigned g = __hip_atomic_load(bgen, __ATOMIC_ACQUIRE,
                                           __HIP_MEMORY_SCOPE_AGENT);
            unsigned old = __hip_atomic_fetch_add(bcnt, 1u, __ATOMIC_ACQ_REL,
                                                  __HIP_MEMORY_SCOPE_AGENT);
            if (old == MEMBERS - 1) {
                __hip_atomic_store(bcnt, 0u, __ATOMIC_RELEASE,
                                   __HIP_MEMORY_SCOPE_AGENT);
                __hip_atomic_fetch_add(bgen, 1u, __ATOMIC_ACQ_REL,
                                       __HIP_MEMORY_SCOPE_AGENT);
            } else {
                while (__hip_atomic_load(bgen, __ATOMIC_ACQUIRE,
                                         __HIP_MEMORY_SCOPE_AGENT) == g) {
                    __builtin_amdgcn_s_sleep(2);
                }
            }
        }
        __syncthreads();
    }
}

// Final readout: out = h2(T-1) @ Wfc + b_fc  (f32 stash).
__global__ void fc_out_kernel(const float* __restrict__ h2fin,
                              const float* __restrict__ Wfc,
                              const float* __restrict__ b_fc,
                              float* __restrict__ out) {
    const int i = blockIdx.x * blockDim.x + threadIdx.x;
    if (i >= B_SZ * C_SZ) return;
    const int b = i / C_SZ, c = i % C_SZ;
    const float* h = h2fin + (size_t)b * N_SZ;
    float acc = b_fc[c];
    for (int k = 0; k < N_SZ; ++k) acc += h[k] * Wfc[(size_t)k * C_SZ + c];
    out[i] = acc;
}

// ---------------------------------------------------------------------------
extern "C" void kernel_launch(void* const* d_in, const int* in_sizes, int n_in,
                              void* d_out, int out_size, void* d_ws, size_t ws_size,
                              hipStream_t stream) {
    const float* data = (const float*)d_in[0];
    const float* h0   = (const float*)d_in[1];
    const float* Win  = (const float*)d_in[2];
    const float* b_in = (const float*)d_in[3];
    const float* Whh  = (const float*)d_in[4];
    const float* b_hh = (const float*)d_in[5];
    const float* Whi  = (const float*)d_in[6];
    const float* b_hi = (const float*)d_in[7];
    const float* Wfc  = (const float*)d_in[8];
    const float* b_fc = (const float*)d_in[9];
    float* out = (float*)d_out;

    char* ws = (char*)d_ws;
    size_t off = 0;
    auto carve = [&](size_t bytes) -> void* {
        void* p = ws + off;
        off += (bytes + 255) & ~(size_t)255;
        return p;
    };
    __bf16*   data_bf = (__bf16*)carve((size_t)T_STEPS * B_SZ * I_SZ * 2);
    __bf16*   pkWin   = (__bf16*)carve((size_t)I_SZ * N_SZ * 2);
    __bf16*   pkWhh   = (__bf16*)carve((size_t)3 * N_SZ * N_SZ * 2);
    __bf16*   pkWhi   = (__bf16*)carve((size_t)2 * N_SZ * N_SZ * 2);
    __bf16*   hbf     = (__bf16*)carve((size_t)2 * 3 * B_SZ * N_SZ * 2);
    float*    h2fin   = (float*)carve((size_t)B_SZ * N_SZ * 4);
    unsigned* bar     = (unsigned*)carve(256);

    const int nd = T_STEPS * B_SZ * I_SZ;
    f32_to_bf16_kernel<<<(nd + 255) / 256, 256, 0, stream>>>(data, data_bf, nd);

    pack_b_kernel<<<(KB_I * NB_N * 32 + 255) / 256, 256, 0, stream>>>(Win, pkWin, I_SZ, 0);
    for (int l = 0; l < 3; ++l)
        pack_b_kernel<<<(KB_N * NB_N * 32 + 255) / 256, 256, 0, stream>>>(
            Whh + (size_t)l * N_SZ * N_SZ, pkWhh + (size_t)l * N_SZ * N_SZ, N_SZ, 1);
    for (int l = 0; l < 2; ++l)
        pack_b_kernel<<<(KB_N * NB_N * 32 + 255) / 256, 256, 0, stream>>>(
            Whi + (size_t)l * N_SZ * N_SZ, pkWhi + (size_t)l * N_SZ * N_SZ, N_SZ, 0);

    const int ns = 3 * B_SZ * N_SZ;
    init_state_kernel<<<(ns + 255) / 256, 256, 0, stream>>>(h0, hbf, bar, ns);

    rnn_stack_kernel<<<dim3(NSPLIT, MTILES, L_SZ), THREADS, LDS_BYTES, stream>>>(
        data_bf, pkWin, pkWhh, pkWhi, h0, b_in, b_hh, b_hi, hbf, h2fin, bar);

    fc_out_kernel<<<(B_SZ * C_SZ + 255) / 256, 256, 0, stream>>>(h2fin, Wfc, b_fc, out);
}
